// GNNForConsensus_89988154786374
// MI455X (gfx1250) — matrix operand order repaired and checked by reading, not compiled, not run on hardware
//
#include <hip/hip_runtime.h>
#include <hip/hip_bf16.h>
#include <stdint.h>

// ---------------------------------------------------------------------------
// GCN 2-layer forward for MI455X (gfx1250, wave32).
//   deg/norm  : f32 atomics (self-loops folded in analytically)
//   transform : v_wmma_f32_16x16x32_f16 (f16 A/B, f32 accum), W in LDS (f16,
//               transposed, pitch 136 halves -> 16B-aligned ds_load_b128).
//               nt-loop kept rolled so B fragments stream from LDS each
//               iteration instead of being hoisted+spilled to scratch.
//   aggregate : wave-per-edge float4 gather (L2-resident: 192MB L2 holds all
//               51MB feature buffers) + non-returning global_atomic_add_f32
// ---------------------------------------------------------------------------

typedef __attribute__((ext_vector_type(16))) _Float16 v16h;
typedef __attribute__((ext_vector_type(8)))  _Float16 v8h;
typedef __attribute__((ext_vector_type(8)))  float    v8f;

#define N_NODES 100000
#define HDIM    128
#define M_TILES (N_NODES / 16)   // 6250, exact

// ---------------- degree / norm kernels ----------------

__global__ void k_deg_init(float* __restrict__ deg, int n) {
    int i = blockIdx.x * blockDim.x + threadIdx.x;
    if (i < n) deg[i] = 1.0f;                    // self-loop contribution
}

__global__ void k_deg_edges(const int* __restrict__ dst, float* __restrict__ deg, int e) {
    int i = blockIdx.x * blockDim.x + threadIdx.x;
    if (i < e) atomicAdd(&deg[dst[i]], 1.0f);
}

__global__ void k_dinv(float* __restrict__ deg, int n) {
    int i = blockIdx.x * blockDim.x + threadIdx.x;
    if (i < n) deg[i] = rsqrtf(deg[i]);          // deg >= 1 always (self-loops)
}

// ---------------- dense transform: H = act(A) @ W via WMMA ----------------
// A: [N,128] f32 row-major.  W: [128,128] f32 row-major (k,n).
// LDS holds W^T as f16: wT[n][k], pitch 136 halves (272B = 16B multiple).

template <bool RELU>
__global__ void __launch_bounds__(128)
k_gemm_wmma(const float* __restrict__ A,
            const float* __restrict__ W,
            float* __restrict__ H,
            int n_tiles) {
    __shared__ __align__(16) _Float16 wT[128 * 136];

    // cooperative load + transpose + f32->f16 (once per workgroup)
    for (int i = threadIdx.x; i < HDIM * HDIM; i += blockDim.x) {
        int k = i >> 7, n = i & 127;
        wT[n * 136 + k] = (_Float16)W[i];
    }
    __syncthreads();

    const int wave   = threadIdx.x >> 5;
    const int lane   = threadIdx.x & 31;
    const int mlane  = lane & 15;        // M (for A/store) == N-local (for B/store)
    const int g      = lane >> 4;        // K-half selector
    const int wstride = gridDim.x * (blockDim.x >> 5);

    for (int tile = blockIdx.x * (blockDim.x >> 5) + wave; tile < n_tiles; tile += wstride) {
        const float* arow = A + (size_t)(tile * 16 + mlane) * HDIM;

        // A fragments: 4 chunks of K=32. ISA layout: lane holds K = 32c+8g+0..7
        // in elems 0..7 and K = 32c+16+8g+0..7 in elems 8..15.
        v16h afrag[4];
#pragma unroll
        for (int c = 0; c < 4; ++c) {
            int k0 = c * 32 + g * 8;
            float4 p0 = *(const float4*)(arow + k0);
            float4 p1 = *(const float4*)(arow + k0 + 4);
            float4 p2 = *(const float4*)(arow + k0 + 16);
            float4 p3 = *(const float4*)(arow + k0 + 20);
            float t[16] = {p0.x, p0.y, p0.z, p0.w, p1.x, p1.y, p1.z, p1.w,
                           p2.x, p2.y, p2.z, p2.w, p3.x, p3.y, p3.z, p3.w};
            v16h a;
#pragma unroll
            for (int j = 0; j < 16; ++j) {
                float v = t[j];
                if (RELU) v = fmaxf(v, 0.0f);   // fused ReLU on layer-2 input
                a[j] = (_Float16)v;
            }
            afrag[c] = a;
        }

        // C-store base: elem r of acc -> row (16*tile + r + 8g), col (16*nt + mlane)
        float* crow = H + (size_t)(tile * 16 + 8 * g) * HDIM + mlane;

        // Keep rolled: B loads must stay inside (depend on nt) so they remain
        // ds_load_b128 from LDS instead of being hoisted and spilled to scratch.
#pragma clang loop unroll(disable)
        for (int nt = 0; nt < 8; ++nt) {
            v8f acc = {};
            const _Float16* bp = &wT[(nt * 16 + mlane) * 136 + g * 8];
#pragma unroll
            for (int c = 0; c < 4; ++c) {
                // B layout: lane holds column n, K = 8g + 0..7 and 16 + 8g + 0..7 (matches A)
                v8h blo = *(const v8h*)(bp + c * 32);
                v8h bhi = *(const v8h*)(bp + c * 32 + 16);
                v16h b = __builtin_shufflevector(blo, bhi, 0, 1, 2, 3, 4, 5, 6, 7,
                                                 8, 9, 10, 11, 12, 13, 14, 15);
                acc = __builtin_amdgcn_wmma_f32_16x16x32_f16(
                    false, afrag[c], false, b, (short)0, acc, false, false);
            }
            float* cp = crow + nt * 16;
#pragma unroll
            for (int r = 0; r < 8; ++r)
                cp[(size_t)r * HDIM] = acc[r];   // constant 512B immediate offsets
        }
    }
}

// ---------------- aggregation ----------------
// out[i] = b + dinv[i]^2 * h[i]   (bias + self-loop, fully overwrites out)
__global__ void k_agg_init(const float* __restrict__ h, const float* __restrict__ dinv,
                           const float* __restrict__ bias, float* __restrict__ out, int n) {
    long long idx = (long long)blockIdx.x * blockDim.x + threadIdx.x;  // n*32 threads
    if (idx >= (long long)n * 32) return;
    int node = (int)(idx >> 5), q = (int)(idx & 31);
    float w = dinv[node] * dinv[node];
    float4 hv = ((const float4*)(h + (size_t)node * HDIM))[q];
    float4 bv = ((const float4*)bias)[q];
    float4 r = {bv.x + w * hv.x, bv.y + w * hv.y, bv.z + w * hv.z, bv.w + w * hv.w};
    ((float4*)(out + (size_t)node * HDIM))[q] = r;
}

// out[dst] += dinv[src]*dinv[dst] * h[src] ; one wave (32 lanes) per edge,
// 4 features/lane. Gathers hit L2 (h is 51MB << 192MB L2); scatter uses
// non-returning global_atomic_add_f32.
__global__ void k_agg_edges(const float* __restrict__ h, const float* __restrict__ dinv,
                            const int* __restrict__ src, const int* __restrict__ dst,
                            float* __restrict__ out, int e) {
    int edge = (int)(((long long)blockIdx.x * blockDim.x + threadIdx.x) >> 5);
    int lane = threadIdx.x & 31;
    if (edge >= e) return;
    int s = src[edge], d = dst[edge];
    float w = dinv[s] * dinv[d];
    float4 v = ((const float4*)(h + (size_t)s * HDIM))[lane];
    float* o = out + (size_t)d * HDIM + lane * 4;
    atomicAdd(o + 0, w * v.x);
    atomicAdd(o + 1, w * v.y);
    atomicAdd(o + 2, w * v.z);
    atomicAdd(o + 3, w * v.w);
}

// ---------------- launch ----------------

extern "C" void kernel_launch(void* const* d_in, const int* in_sizes, int n_in,
                              void* d_out, int out_size, void* d_ws, size_t ws_size,
                              hipStream_t stream) {
    const float* x    = (const float*)d_in[0];
    const int*   ei   = (const int*)d_in[1];       // [2,E] int32 (JAX x64 off)
    const float* W1   = (const float*)d_in[2];
    const float* b1   = (const float*)d_in[3];
    const float* W2   = (const float*)d_in[4];
    const float* b2   = (const float*)d_in[5];
    float*       out  = (float*)d_out;

    const int N = N_NODES;
    const int E = in_sizes[1] / 2;
    const int* src = ei;
    const int* dst = ei + E;

    // workspace: dinv[N] | h[N*128] | agg[N*128]   (~103 MB)
    float* dinv = (float*)d_ws;
    float* h    = dinv + ((N + 255) & ~255);       // keep 16B alignment
    float* agg  = h + (size_t)N * HDIM;

    const int B = 256;
    // degrees -> dinv
    k_deg_init<<<(N + B - 1) / B, B, 0, stream>>>(dinv, N);
    k_deg_edges<<<(E + B - 1) / B, B, 0, stream>>>(dst, dinv, E);
    k_dinv<<<(N + B - 1) / B, B, 0, stream>>>(dinv, N);

    long long initThreads = (long long)N * 32;
    int initBlocks = (int)((initThreads + B - 1) / B);
    int edgeBlocks = (int)(((long long)E * 32 + B - 1) / B);

    // layer 1: h = x @ W1 ; agg = b1 + self + edge-sum
    k_gemm_wmma<false><<<256, 128, 0, stream>>>(x, W1, h, M_TILES);
    k_agg_init<<<initBlocks, B, 0, stream>>>(h, dinv, b1, agg, N);
    k_agg_edges<<<edgeBlocks, B, 0, stream>>>(h, dinv, src, dst, agg, E);

    // layer 2: ReLU fused into gemm input load; h reused; result -> d_out
    k_gemm_wmma<true><<<256, 128, 0, stream>>>(agg, W2, h, M_TILES);
    k_agg_init<<<initBlocks, B, 0, stream>>>(h, dinv, b2, out, N);
    k_agg_edges<<<edgeBlocks, B, 0, stream>>>(h, dinv, src, dst, out, E);
}
